// VectorQuantizer_9079560863775
// MI455X (gfx1250) — matrix operand order repaired and checked by reading, not compiled, not run on hardware
//
#include <hip/hip_runtime.h>
#include <hip/hip_bf16.h>

// Problem constants (reference: inputs [16,2048,256] f32, weight [1024,256] f32)
#define BT       32768      // 16*2048 tokens
#define DIM      256
#define KCODES   1024
#define MTILE    128        // tokens per block
#define NTHREADS 256        // 8 waves of 32
#define ROWB     528        // LDS x row stride bytes (256*2 + 16 pad)
#define BROWB    528        // LDS B row stride bytes (256*2 + 16 pad)
#define BBUF     (16 * BROWB)

// Dynamic LDS layout
#define XS_OFF   0
#define XS_SIZE  (MTILE * ROWB)            // 67584
#define B_OFF    XS_SIZE                   // 2 x 8448 double buffer
#define IDX_OFF  (B_OFF + 2 * BBUF)        // 84480
#define RED_OFF  (IDX_OFF + MTILE * 4)     // 84992
#define SMEM_BYTES (RED_OFF + NTHREADS * 4) // 86016

// Output layout in d_out (floats), reference return order:
// quantized_st[BT*D], indices[BT], q_loss, e_loss, perplexity, encodings[BT*K]
#define OFF_Q    0
#define OFF_IDX  (BT * DIM)
#define OFF_QL   (OFF_IDX + BT)
#define OFF_EL   (OFF_QL + 1)
#define OFF_PP   (OFF_EL + 1)
#define OFF_ENC  (OFF_PP + 1)              // odd float offset -> b32 stores

typedef __attribute__((ext_vector_type(16))) __bf16 v16bf;
typedef __attribute__((ext_vector_type(8)))  float  v8f;

__device__ __forceinline__ unsigned short f2bf(float f) {
  unsigned u = __float_as_uint(f);
  u += 0x7fffu + ((u >> 16) & 1u);         // round-to-nearest-even
  return (unsigned short)(u >> 16);
}

// ---- CDNA5 async global->LDS copy (ISA 15.18.3 op 98, tracked by ASYNCcnt) ----
__device__ __forceinline__ unsigned lds_off(const void* p) {
  // flat shared addr low 32 bits == LDS byte address (ISA 10.2 aperture rules)
  return (unsigned)(unsigned long long)p;
}
__device__ __forceinline__ void async_b128(unsigned ldsByteOff, const void* g) {
  asm volatile("global_load_async_to_lds_b128 %0, %1, off"
               :: "v"(ldsByteOff), "v"((unsigned long long)g)
               : "memory");
}
__device__ __forceinline__ void wait_async0() {
  asm volatile("s_wait_asynccnt 0x0" ::: "memory");
}

// ---------------------------------------------------------------- init
__global__ void vq_init(float* acc) {
  if (threadIdx.x < 2) acc[threadIdx.x] = 0.0f;
}

// ------------------------------------------------- codebook L2-normalize -> bf16
__global__ __launch_bounds__(256)
void vq_norm_weight(const float* __restrict__ w, unsigned short* __restrict__ wbf) {
  __shared__ float red[256];
  const int row = blockIdx.x;
  const int c   = threadIdx.x;
  float v = w[row * DIM + c];
  red[c] = v * v;
  __syncthreads();
  #pragma unroll
  for (int s = 128; s > 0; s >>= 1) {
    if (c < s) red[c] += red[c + s];
    __syncthreads();
  }
  float inv = 1.0f / fmaxf(sqrtf(red[0]), 1e-12f);
  wbf[row * DIM + c] = f2bf(v * inv);
}

// ---------------------------------------------------------------- main
// scores[m,n] = sum_d x[m,d] * w_hat[n,d]; x-normalization skipped (argmax-invariant).
__global__ __launch_bounds__(NTHREADS)
void vq_main(const float* __restrict__ x, const float* __restrict__ w,
             const unsigned short* __restrict__ wbf,
             float* __restrict__ out, float* __restrict__ acc) {
  extern __shared__ unsigned char smem[];
  unsigned char* xs      = smem + XS_OFF;
  unsigned char* bb      = smem + B_OFF;
  int*           lds_idx = (int*)(smem + IDX_OFF);
  float*         red     = (float*)(smem + RED_OFF);

  const int  tid  = threadIdx.x;
  const int  wv   = tid >> 5;       // wave 0..7, owns rows wv*16..wv*16+15
  const int  lane = tid & 31;
  const int  lrow = lane & 15;
  const int  half = lane >> 4;
  const long rowBase = (long)blockIdx.x * MTILE;

  // ---- Stage A: global f32 x-tile -> LDS bf16 (128 rows x 256)
  #pragma unroll 4
  for (int j = 0; j < 32; ++j) {
    int e  = tid + j * NTHREADS;          // float4 index, 8192 total
    int r  = e >> 6;
    int c4 = e & 63;
    float4 f = ((const float4*)x)[rowBase * (DIM / 4) + e];
    uint2 pk;
    pk.x = (unsigned)f2bf(f.x) | ((unsigned)f2bf(f.y) << 16);
    pk.y = (unsigned)f2bf(f.z) | ((unsigned)f2bf(f.w) << 16);
    *(uint2*)(xs + r * ROWB + c4 * 8) = pk;
  }

  // ---- Prologue: async-stage B tile 0 into buffer 0 (16 codes x 512B, padded rows)
  #pragma unroll
  for (int q = 0; q < 2; ++q) {
    int chunk = tid + q * NTHREADS;       // 512 b128 chunks
    int brow  = chunk >> 5;
    int c16   = chunk & 31;
    async_b128(lds_off(bb + brow * BROWB + c16 * 16),
               wbf + brow * DIM + c16 * 8);
  }
  wait_async0();
  __syncthreads();

  // ---- Preload A fragments (ISA 16-bit A 16x32 layout):
  // lanes 0-15: row=lrow, K = s*32 + {0..7,16..23}; lanes 16-31: K = s*32 + {8..15,24..31}
  v16bf a[8];
  {
    const unsigned char* abase = xs + (wv * 16 + lrow) * ROWB + half * 16;
    #pragma unroll
    for (int s = 0; s < 8; ++s) {
      float4* p = (float4*)&a[s];
      p[0] = *(const float4*)(abase + s * 64);
      p[1] = *(const float4*)(abase + s * 64 + 32);
    }
  }

  float best[8];
  int   bidx[8];
  #pragma unroll
  for (int r = 0; r < 8; ++r) { best[r] = -3.4e38f; bidx[r] = 0; }

  // ---- 64 n-tiles; double-buffered async B staging overlapped with 8 WMMAs
  for (int nt = 0; nt < 64; ++nt) {
    const unsigned char* bcur = bb + (nt & 1) * BBUF;

    if (nt + 1 < 64) {                    // issue async copy of next tile
      unsigned char* bnxt = bb + ((nt + 1) & 1) * BBUF;
      #pragma unroll
      for (int q = 0; q < 2; ++q) {
        int chunk = tid + q * NTHREADS;
        int brow  = chunk >> 5;
        int c16   = chunk & 31;
        async_b128(lds_off(bnxt + brow * BROWB + c16 * 16),
                   wbf + ((nt + 1) * 16 + brow) * DIM + c16 * 8);
      }
    }

    // B fragment (32x16 bf16): lane holds d = {0..7,16..23} of code n=lrow;
    // lanes 16-31 hold d = {8..15,24..31} (same K order as A).
    const unsigned char* bfrag = bcur + lrow * BROWB + half * 16;
    v8f c = {0.f, 0.f, 0.f, 0.f, 0.f, 0.f, 0.f, 0.f};
    #pragma unroll
    for (int s = 0; s < 8; ++s) {
      v16bf b;
      float4* p = (float4*)&b;
      p[0] = *(const float4*)(bfrag + s * 64);
      p[1] = *(const float4*)(bfrag + s * 64 + 32);
      c = __builtin_amdgcn_wmma_f32_16x16x32_bf16(
              false, a[s], false, b, (short)0, c, false, false);
    }

    const int ncode = nt * 16 + lrow;
    #pragma unroll
    for (int r = 0; r < 8; ++r) {
      if (c[r] > best[r]) { best[r] = c[r]; bidx[r] = ncode; }
    }

    wait_async0();      // next tile landed in LDS
    __syncthreads();    // publish to all waves / protect buffer reuse
  }

  // ---- cross-lane argmax within each 16-lane half (C layout: slot r ->
  // M=r for lanes 0-15, M=8+r for lanes 16-31; N=lane&15)
  #pragma unroll
  for (int r = 0; r < 8; ++r) {
    float v = best[r];
    int   i = bidx[r];
    #pragma unroll
    for (int m = 8; m >= 1; m >>= 1) {
      float ov = __shfl_xor(v, m, 32);
      int   oi = __shfl_xor(i, m, 32);
      if (ov > v || (ov == v && oi < i)) { v = ov; i = oi; }  // first-match tie-break
    }
    if (lrow == 0) {
      int row = wv * 16 + half * 8 + r;
      lds_idx[row] = i;
      out[OFF_IDX + rowBase + row] = (float)i;
    }
  }
  __syncthreads();

  // ---- Stage C: gather quantized = weight[idx], write out, accumulate MSE
  float lacc = 0.0f;
  #pragma unroll 4
  for (int j = 0; j < 32; ++j) {
    int e  = tid + j * NTHREADS;
    int r  = e >> 6;
    int c4 = e & 63;
    int idx = lds_idx[r];
    float4 q  = ((const float4*)w)[idx * (DIM / 4) + c4];
    float4 xv = ((const float4*)x)[(rowBase + r) * (DIM / 4) + c4];
    ((float4*)out)[(rowBase + r) * (DIM / 4) + c4] = q;   // quantized_st == quantized
    float dx = q.x - xv.x, dy = q.y - xv.y, dz = q.z - xv.z, dw = q.w - xv.w;
    lacc += dx * dx + dy * dy + dz * dz + dw * dw;
  }
  red[tid] = lacc;
  __syncthreads();
  #pragma unroll
  for (int s = NTHREADS / 2; s > 0; s >>= 1) {
    if (tid < s) red[tid] += red[tid + s];
    __syncthreads();
  }
  if (tid == 0) atomicAdd(&acc[0], red[0]);

  // ---- Stage D: one-hot encodings (odd base offset -> coalesced b32 stores)
  for (int j = 0; j < 512; ++j) {
    int e = tid + j * NTHREADS;       // 131072 = 128 rows * 1024
    int r = e >> 10;
    int k = e & 1023;
    out[OFF_ENC + (rowBase + r) * KCODES + k] = (lds_idx[r] == k) ? 1.0f : 0.0f;
  }
}

// ------------------------------------------------- perplexity over avg_probs[T,K]
__global__ __launch_bounds__(256)
void vq_perplexity(const float* __restrict__ outIdx, float* __restrict__ acc) {
  int t = blockIdx.x * blockDim.x + threadIdx.x;
  if (t >= 2048) return;
  int ids[16];
  #pragma unroll
  for (int b = 0; b < 16; ++b) ids[b] = (int)outIdx[b * 2048 + t];
  float s = 0.0f;
  #pragma unroll
  for (int i = 0; i < 16; ++i) {
    bool first = true;
    for (int j = 0; j < i; ++j)
      if (ids[j] == ids[i]) { first = false; break; }
    if (!first) continue;
    int c = 1;
    for (int j = i + 1; j < 16; ++j)
      if (ids[j] == ids[i]) ++c;
    float p = (float)c * (1.0f / 16.0f);
    s += p * logf(p + 1e-10f);
  }
  atomicAdd(&acc[1], s);
}

// ---------------------------------------------------------------- finalize
__global__ void vq_finalize(const float* __restrict__ acc, float* __restrict__ out) {
  if (threadIdx.x == 0) {
    float q = acc[0] * (1.0f / (float)(BT * DIM));
    out[OFF_QL] = q;            // q_latent_loss
    out[OFF_EL] = 0.25f * q;    // e_latent_loss = commitment * same mean
    out[OFF_PP] = expf(-acc[1]);
  }
}

extern "C" void kernel_launch(void* const* d_in, const int* in_sizes, int n_in,
                              void* d_out, int out_size, void* d_ws, size_t ws_size,
                              hipStream_t stream) {
  const float* x = (const float*)d_in[0];   // [16,2048,256] f32
  const float* w = (const float*)d_in[1];   // [1024,256] f32
  float* out = (float*)d_out;
  float* acc = (float*)d_ws;                                  // 2 f32 accumulators
  unsigned short* wbf = (unsigned short*)((char*)d_ws + 64);  // bf16 codebook, 512KB

  vq_init<<<1, 32, 0, stream>>>(acc);
  vq_norm_weight<<<KCODES, 256, 0, stream>>>(w, wbf);
  vq_main<<<BT / MTILE, NTHREADS, SMEM_BYTES, stream>>>(x, w, wbf, out, acc);
  vq_perplexity<<<2048 / 256, 256, 0, stream>>>(out + OFF_IDX, acc);
  vq_finalize<<<1, 32, 0, stream>>>(acc, out);
}
